// EEG3DDeformAutoencoder_37099927503208
// MI455X (gfx1250) — matrix operand order repaired and checked by reading, not compiled, not run on hardware
//
#include <hip/hip_runtime.h>
#include <hip/hip_bf16.h>
#include <math.h>

// ---------------------------------------------------------------------------
// EEG3DDeformAutoencoder on MI455X (gfx1250, wave32, WMMA)
//
// Pipeline (all intermediates channel-last so gathers read 32 contiguous ch):
//   K0  deform1 fused (offsets+gather+27->32 einsum)        -> h1_pre
//   S   channel stats (sum,sumsq)                           (generic)
//   K2  BN+ELU+maxpool2                                     -> h1p
//   K3a offset GEMM  M=73728 N=96(81) K=864   [WMMA f16]    -> off2
//   K3c deform GEMM  M=73728 N=64    K=864    [WMMA f16]    -> h2_pre
//       (A-tile trilinear-gathered into LDS per K-step; off2 slab staged
//        into LDS via GLOBAL_LOAD_ASYNC_TO_LDS, tracked by ASYNCcnt)
//   S   stats ; K5 BN+ELU+maxpool2 -> codes (d_out, NCDHW)
//   K6  2x2x2 mean + FC -> classes (d_out)
//   K7  convT 64->32 ; stats ; BN+ELU ; K10 convT 32->1 ; stats ; BN+ELU -> d
// ---------------------------------------------------------------------------

typedef __attribute__((ext_vector_type(16))) _Float16 v16h;
typedef __attribute__((ext_vector_type(8)))  float    v8f;

#define B_     8
#define D1_    24
#define H1_    48
#define W1_    64
#define SP1    73728        // 24*48*64
#define NV1    589824       // 8*SP1
#define D2_    12
#define H2_    24
#define W2_    32
#define SP2    9216         // 12*24*32
#define NR2    73728        // 8*SP2

// workspace float offsets (with reuse; peak ~85 MB)
#define OFF_H1PRE 0u                 // 18,874,368 f   (dead after K2)
#define OFF_OFF2  0u                 //  7,077,888 f   (reuses h1_pre)
#define OFF_H2    7077888u           //  4,718,592 f
#define OFF_D1P   0u                 //  2,359,296 f   (reuses off2 after K3c/K5)
#define OFF_D1A   2359296u           //  2,359,296 f
#define OFF_D2P   4718592u           //    589,824 f
#define OFF_H1P   18874368u          //  2,359,296 f
#define OFF_STATS 21233664u          //        512 f
#define WS_FLOATS (OFF_STATS + 512u)

#ifndef __has_builtin
#define __has_builtin(x) 0
#endif

// ---------------------------------------------------------------------------
// helpers
// ---------------------------------------------------------------------------
__device__ __forceinline__ float fetch_pad1(const float* xb, int qd, int qh, int qw,
                                            int D, int H, int W) {
  // padded coord q (pad=1) -> raw coord q-1, zero outside
  int dd = qd - 1, hh = qh - 1, ww = qw - 1;
  return ((unsigned)dd < (unsigned)D && (unsigned)hh < (unsigned)H && (unsigned)ww < (unsigned)W)
             ? xb[((size_t)dd * H + hh) * W + ww]
             : 0.f;
}

// A-fragment addressing for V_WMMA f16 16x16x32 (ISA 7.12.2):
// lane = m + 16*half, half = (k>>3)&1, element j = (k&7) + (k>=16 ? 8 : 0)
__device__ __forceinline__ int a_frag_index(int m_local, int k_local) {
  int half = (k_local >> 3) & 1;
  int j    = (k_local & 7) + ((k_local >> 4) << 3);
  return (m_local + 16 * half) * 16 + j;
}

// ---------------------------------------------------------------------------
// K0: fused deformable conv layer 1 (C_in=1 -> 32), scalar VALU (gather bound)
// h1 layout: channel-last rows [b*SP1 + sp][32]
// ---------------------------------------------------------------------------
__global__ void deform1_kernel(const float* __restrict__ x, const float* __restrict__ w_off1,
                               const float* __restrict__ b_off1, const float* __restrict__ w1,
                               const float* __restrict__ b1, float* __restrict__ h1) {
  int i = blockIdx.x * 256 + threadIdx.x;        // exactly NV1 threads
  int w = i & 63;
  int t1 = i >> 6;
  int h = t1 % H1_;
  int t2 = t1 / H1_;
  int d = t2 % D1_;
  int b = t2 / D1_;
  const float* xb = x + (size_t)b * SP1;

  // 27 input taps (zero padded) kept in registers
  float xv[27];
#pragma unroll
  for (int j = 0; j < 27; ++j) {
    int kd = j / 9 - 1, kh = (j / 3) % 3 - 1, kw = j % 3 - 1;
    int dz = d + kd, hy = h + kh, wx = w + kw;
    xv[j] = ((unsigned)dz < (unsigned)D1_ && (unsigned)hy < (unsigned)H1_ &&
             (unsigned)wx < (unsigned)W1_)
                ? xb[((size_t)dz * H1_ + hy) * W1_ + wx]
                : 0.f;
  }

  float v[27];
  for (int t = 0; t < 27; ++t) {
    float od = b_off1[t], oh = b_off1[27 + t], ow = b_off1[54 + t];
#pragma unroll
    for (int j = 0; j < 27; ++j) {
      od += xv[j] * w_off1[t * 27 + j];
      oh += xv[j] * w_off1[(27 + t) * 27 + j];
      ow += xv[j] * w_off1[(54 + t) * 27 + j];
    }
    int td = t / 9 - 1, th = (t / 3) % 3 - 1, tw = t % 3 - 1;
    float pd = fminf(fmaxf((float)(d + 1 + td) + od, 0.f), (float)(D1_ + 1));
    float ph = fminf(fmaxf((float)(h + 1 + th) + oh, 0.f), (float)(H1_ + 1));
    float pw = fminf(fmaxf((float)(w + 1 + tw) + ow, 0.f), (float)(W1_ + 1));
    float d0f = floorf(pd), h0f = floorf(ph), w0f = floorf(pw);
    int d0 = (int)d0f, h0 = (int)h0f, w0 = (int)w0f;
    int d1 = min(d0 + 1, D1_ + 1), hh1 = min(h0 + 1, H1_ + 1), w1c = min(w0 + 1, W1_ + 1);
    float fd = pd - d0f, fh = ph - h0f, fw = pw - w0f;
    float c000 = fetch_pad1(xb, d0, h0, w0, D1_, H1_, W1_);
    float c001 = fetch_pad1(xb, d0, h0, w1c, D1_, H1_, W1_);
    float c010 = fetch_pad1(xb, d0, hh1, w0, D1_, H1_, W1_);
    float c011 = fetch_pad1(xb, d0, hh1, w1c, D1_, H1_, W1_);
    float c100 = fetch_pad1(xb, d1, h0, w0, D1_, H1_, W1_);
    float c101 = fetch_pad1(xb, d1, h0, w1c, D1_, H1_, W1_);
    float c110 = fetch_pad1(xb, d1, hh1, w0, D1_, H1_, W1_);
    float c111 = fetch_pad1(xb, d1, hh1, w1c, D1_, H1_, W1_);
    float lo = (c000 * (1.f - fw) + c001 * fw) * (1.f - fh) + (c010 * (1.f - fw) + c011 * fw) * fh;
    float hi = (c100 * (1.f - fw) + c101 * fw) * (1.f - fh) + (c110 * (1.f - fw) + c111 * fw) * fh;
    v[t] = lo * (1.f - fd) + hi * fd;
  }

  float* orow = h1 + (size_t)i * 32;
  for (int o = 0; o < 32; ++o) {
    float acc = b1[o];
#pragma unroll
    for (int t = 0; t < 27; ++t) acc += v[t] * w1[o * 27 + t];
    orow[o] = acc;
  }
}

// ---------------------------------------------------------------------------
// generic channel stats (channel-last): out[0..C-1]=sum, out[C..2C-1]=sumsq
// ---------------------------------------------------------------------------
__global__ void stats_kernel(const float* __restrict__ x, int rows, int C,
                             float* __restrict__ out) {
  __shared__ float ssum[64], ssq[64];
  int tid = threadIdx.x;
  if (tid < C) { ssum[tid] = 0.f; ssq[tid] = 0.f; }
  __syncthreads();
  int tpr = 256 / C;               // C in {1,32,64}
  int c = tid % C;
  int rl = tid / C;
  float s = 0.f, q = 0.f;
  for (long long r = (long long)blockIdx.x * tpr + rl; r < rows;
       r += (long long)gridDim.x * tpr) {
    float v = x[r * C + c];
    s += v;
    q += v * v;
  }
  atomicAdd(&ssum[c], s);
  atomicAdd(&ssq[c], q);
  __syncthreads();
  if (tid < C) {
    atomicAdd(&out[tid], ssum[tid]);
    atomicAdd(&out[C + tid], ssq[tid]);
  }
}

// ---------------------------------------------------------------------------
// BN + ELU + 2x2x2 maxpool (channel-last in; channel-last and/or NCDHW out)
// ---------------------------------------------------------------------------
__global__ void bn_elu_pool_kernel(const float* __restrict__ xp, float* __restrict__ out_cl,
                                   float* __restrict__ out_nchw, int Dd, int Hh, int Ww, int C,
                                   const float* __restrict__ stats, const float* __restrict__ g,
                                   const float* __restrict__ be, float invn) {
  int i = blockIdx.x * 256 + threadIdx.x;
  int Pd = Dd >> 1, Ph = Hh >> 1, Pw = Ww >> 1;
  int c = i % C;
  int t = i / C;
  int pw = t % Pw; t /= Pw;
  int ph = t % Ph; t /= Ph;
  int pd = t % Pd;
  int b = t / Pd;
  float m = stats[c] * invn;
  float var = stats[C + c] * invn - m * m;
  float sc = g[c] * rsqrtf(var + 1e-5f);
  float sh = be[c] - m * sc;
  float best = -3.4e38f;
#pragma unroll
  for (int dz = 0; dz < 2; ++dz)
#pragma unroll
    for (int hy = 0; hy < 2; ++hy)
#pragma unroll
      for (int wx = 0; wx < 2; ++wx) {
        float xv = xp[((((size_t)b * Dd + (2 * pd + dz)) * Hh + (2 * ph + hy)) * Ww +
                       (2 * pw + wx)) * C + c];
        float tt = sc * xv + sh;
        tt = tt > 0.f ? tt : expm1f(tt);
        best = fmaxf(best, tt);
      }
  if (out_cl) out_cl[i] = best;
  if (out_nchw)
    out_nchw[((((size_t)b * C + c) * Pd + pd) * Ph + ph) * Pw + pw] = best;
}

// BN + ELU elementwise (channel-last)
__global__ void bn_elu_kernel(const float* __restrict__ xp, float* __restrict__ y, int n, int C,
                              const float* __restrict__ stats, const float* __restrict__ g,
                              const float* __restrict__ be, float invn) {
  int i = blockIdx.x * 256 + threadIdx.x;
  if (i >= n) return;
  int c = i % C;
  float m = stats[c] * invn;
  float var = stats[C + c] * invn - m * m;
  float sc = g[c] * rsqrtf(var + 1e-5f);
  float t = sc * (xp[i] - m) + be[c];
  y[i] = t > 0.f ? t : expm1f(t);
}

// ---------------------------------------------------------------------------
// K3a: layer-2 offset conv as implicit-im2col WMMA GEMM
//      A: M=16 voxels x K=32 (k = tap*32 + c, tap == kt), B: w_off2 (K x 96)
// ---------------------------------------------------------------------------
__global__ void offgemm2_kernel(const float* __restrict__ h1p, const float* __restrict__ w_off2,
                                const float* __restrict__ b_off2, float* __restrict__ off2) {
  __shared__ __align__(32) _Float16 Ash[512];        // 16m x 32k, fragment-linear
  __shared__ __align__(32) _Float16 Bsh[6][512];     // 6 n-tiles of 32k x 16n
  const int tid = threadIdx.x, lane = tid & 31, wv = tid >> 5;
  const int M0 = blockIdx.x * 16;

  const int m_idx = tid >> 4;          // 0..15
  const int c2 = (tid & 15) * 2;       // 2 channels per thread
  const int row = M0 + m_idx;
  const int b = row / SP2;
  int sp = row % SP2;
  const int d = sp / (H2_ * W2_);
  const int h = (sp / W2_) % H2_;
  const int w = sp % W2_;

  v8f acc = {};
  for (int kt = 0; kt < 27; ++kt) {
    // stage B (3072 halfs)
    for (int i2 = tid; i2 < 3072; i2 += 256) {
      int nt = i2 >> 9, ln = (i2 >> 4) & 31, j = i2 & 15;
      int c = (ln >> 4) * 8 + (j & 7) + ((j >> 3) << 4);      // k_local == channel
      int n = nt * 16 + (ln & 15);
      float wv2 = (n < 81) ? w_off2[((size_t)n * 32 + c) * 27 + kt] : 0.f;
      Bsh[nt][ln * 16 + j] = (_Float16)wv2;
    }
    // stage A: im2col tap kt (zero padded), channel-last so reads contiguous
    int td = kt / 9 - 1, th = (kt / 3) % 3 - 1, tw = kt % 3 - 1;
    int dz = d + td, hy = h + th, wx = w + tw;
    bool inb = ((unsigned)dz < (unsigned)D2_) && ((unsigned)hy < (unsigned)H2_) &&
               ((unsigned)wx < (unsigned)W2_);
    const float* src =
        inb ? h1p + ((size_t)b * SP2 + ((dz * H2_ + hy) * W2_ + wx)) * 32 : nullptr;
#pragma unroll
    for (int cc = 0; cc < 2; ++cc) {
      int c = c2 + cc;
      float val = inb ? src[c] : 0.f;
      Ash[a_frag_index(m_idx, c)] = (_Float16)val;
    }
    __syncthreads();
    if (wv < 6) {
      v16h a = *reinterpret_cast<const v16h*>(&Ash[lane * 16]);
      v16h bb = *reinterpret_cast<const v16h*>(&Bsh[wv][lane * 16]);
      acc = __builtin_amdgcn_wmma_f32_16x16x32_f16(false, a, false, bb, (short)0, acc,
                                                   false, false);
    }
    __syncthreads();
  }
  if (wv < 6) {
    int nl = lane & 15, mb = (lane >> 4) * 8;
#pragma unroll
    for (int r = 0; r < 8; ++r) {
      int n = wv * 16 + nl;
      if (n < 81) off2[(size_t)(M0 + mb + r) * 96 + n] = acc[r] + b_off2[n];
    }
  }
}

// ---------------------------------------------------------------------------
// K3c: layer-2 deform einsum as WMMA GEMM with fused trilinear gather.
//      Block = 32 voxels x 64 out-ch; 8 waves = 2 m-subtiles x 4 n-tiles.
//      The block's 32x96 offset slab is staged global->LDS with the async
//      data mover path (ASYNCcnt) when the toolchain exposes the builtins.
// ---------------------------------------------------------------------------
__global__ void deformgemm2_kernel(const float* __restrict__ h1p, const float* __restrict__ off2,
                                   const float* __restrict__ w2, const float* __restrict__ b2,
                                   float* __restrict__ h2) {
  __shared__ __align__(32) _Float16 Ash[2][512];
  __shared__ __align__(32) _Float16 Bsh[4][512];
  __shared__ __align__(16) float OffSh[32 * 96];     // block's offset slab (12 KB)
  const int tid = threadIdx.x, lane = tid & 31, wv = tid >> 5;
  const int mi = wv >> 2, nt = wv & 3;
  const int M0 = blockIdx.x * 32;

  const int m_idx = tid >> 3;          // 0..31
  const int cgrp = (tid & 7) * 4;      // 4 channels per thread
  const int row = M0 + m_idx;
  const int b = row / SP2;
  int sp = row % SP2;
  const int d = sp / (H2_ * W2_);
  const int h = (sp / W2_) % H2_;
  const int w = sp % W2_;
  const float* xb = h1p + (size_t)b * SP2 * 32;

  // ---- stage all 32 rows x 96 offsets into LDS ----
  const float* off_src = off2 + (size_t)M0 * 96;
#if __has_builtin(__builtin_amdgcn_global_load_async_to_lds_b128)
  {
    // async DMA path: 3072 dwords = 768 x b128, 3 per thread; tracked by ASYNCcnt
    typedef int v4i_ __attribute__((vector_size(16)));
    typedef __attribute__((address_space(1))) v4i_* gv4p;
    typedef __attribute__((address_space(3))) v4i_* lv4p;
    for (int i2 = tid * 4; i2 < 32 * 96; i2 += 256 * 4) {
      __builtin_amdgcn_global_load_async_to_lds_b128((gv4p)(void*)(off_src + i2),
                                                     (lv4p)(void*)&OffSh[i2], 0, 0);
    }
#if __has_builtin(__builtin_amdgcn_s_wait_asynccnt)
    __builtin_amdgcn_s_wait_asynccnt(0);
#else
    asm volatile("s_wait_asynccnt 0x0" ::: "memory");
#endif
  }
#else
  for (int i2 = tid; i2 < 32 * 96; i2 += 256) OffSh[i2] = off_src[i2];
#endif
  __syncthreads();
  const float* orow = &OffSh[(m_idx)*96];

  const int mlane = m_idx & 15, msub = m_idx >> 4;
  v8f acc = {};

  for (int kt = 0; kt < 27; ++kt) {
    // stage B (2048 halfs) from w2[o][c][t], t == kt
    for (int i2 = tid; i2 < 2048; i2 += 256) {
      int ntb = i2 >> 9, ln = (i2 >> 4) & 31, j = i2 & 15;
      int c = (ln >> 4) * 8 + (j & 7) + ((j >> 3) << 4);
      int n = ntb * 16 + (ln & 15);
      Bsh[ntb][ln * 16 + j] = (_Float16)w2[((size_t)n * 32 + c) * 27 + kt];
    }
    // stage A: trilinear gather of tap kt for 4 channels
    int td = kt / 9 - 1, th = (kt / 3) % 3 - 1, tw = kt % 3 - 1;
    float od = orow[kt], oh = orow[27 + kt], ow = orow[54 + kt];
    float pd = fminf(fmaxf((float)(d + 1 + td) + od, 0.f), (float)(D2_ + 1));
    float ph = fminf(fmaxf((float)(h + 1 + th) + oh, 0.f), (float)(H2_ + 1));
    float pw = fminf(fmaxf((float)(w + 1 + tw) + ow, 0.f), (float)(W2_ + 1));
    float d0f = floorf(pd), h0f = floorf(ph), w0f = floorf(pw);
    int d0 = (int)d0f, h0 = (int)h0f, w0 = (int)w0f;
    int d1c = min(d0 + 1, D2_ + 1), h1c = min(h0 + 1, H2_ + 1), w1c = min(w0 + 1, W2_ + 1);
    float fd = pd - d0f, fh = ph - h0f, fw = pw - w0f;

    auto cp = [&](int qd, int qh, int qw) -> const float* {
      int dd = qd - 1, hh = qh - 1, ww = qw - 1;
      return ((unsigned)dd < (unsigned)D2_ && (unsigned)hh < (unsigned)H2_ &&
              (unsigned)ww < (unsigned)W2_)
                 ? xb + ((size_t)(dd * H2_ + hh) * W2_ + ww) * 32
                 : nullptr;
    };
    const float* p000 = cp(d0, h0, w0);   const float* p001 = cp(d0, h0, w1c);
    const float* p010 = cp(d0, h1c, w0);  const float* p011 = cp(d0, h1c, w1c);
    const float* p100 = cp(d1c, h0, w0);  const float* p101 = cp(d1c, h0, w1c);
    const float* p110 = cp(d1c, h1c, w0); const float* p111 = cp(d1c, h1c, w1c);
    float g000 = (1.f - fd) * (1.f - fh) * (1.f - fw);
    float g001 = (1.f - fd) * (1.f - fh) * fw;
    float g010 = (1.f - fd) * fh * (1.f - fw);
    float g011 = (1.f - fd) * fh * fw;
    float g100 = fd * (1.f - fh) * (1.f - fw);
    float g101 = fd * (1.f - fh) * fw;
    float g110 = fd * fh * (1.f - fw);
    float g111 = fd * fh * fw;
#pragma unroll
    for (int cc = 0; cc < 4; ++cc) {
      int c = cgrp + cc;
      float val = 0.f;
      if (p000) val += g000 * p000[c];
      if (p001) val += g001 * p001[c];
      if (p010) val += g010 * p010[c];
      if (p011) val += g011 * p011[c];
      if (p100) val += g100 * p100[c];
      if (p101) val += g101 * p101[c];
      if (p110) val += g110 * p110[c];
      if (p111) val += g111 * p111[c];
      Ash[msub][a_frag_index(mlane, c)] = (_Float16)val;
    }
    __syncthreads();
    v16h a = *reinterpret_cast<const v16h*>(&Ash[mi][lane * 16]);
    v16h bb = *reinterpret_cast<const v16h*>(&Bsh[nt][lane * 16]);
    acc = __builtin_amdgcn_wmma_f32_16x16x32_f16(false, a, false, bb, (short)0, acc, false,
                                                 false);
    __syncthreads();
  }
  int nl = lane & 15, mb = (lane >> 4) * 8;
#pragma unroll
  for (int r = 0; r < 8; ++r) {
    int rr = M0 + mi * 16 + mb + r;
    int n = nt * 16 + nl;
    h2[(size_t)rr * 64 + n] = acc[r] + b2[n];
  }
}

// ---------------------------------------------------------------------------
// K6: 2x2x2 mean pooling of codes (NCDHW in d_out) + FC -> classes
// ---------------------------------------------------------------------------
__global__ void fc_kernel(const float* __restrict__ codes, const float* __restrict__ wfc,
                          const float* __restrict__ bfc, float* __restrict__ out) {
  __shared__ float s0[256], s1[256];
  int b = blockIdx.x, tid = threadIdx.x;
  float a0 = 0.f, a1 = 0.f;
  for (int flat = tid; flat < 9216; flat += 256) {
    int c = flat / 144;
    int rem = flat % 144;
    int dd = rem / 48, hh = (rem / 8) % 6, ww = rem % 8;
    float mval = 0.f;
#pragma unroll
    for (int dz = 0; dz < 2; ++dz)
#pragma unroll
      for (int hy = 0; hy < 2; ++hy)
#pragma unroll
        for (int wx = 0; wx < 2; ++wx)
          mval += codes[((((size_t)b * 64 + c) * 6 + (2 * dd + dz)) * 12 + (2 * hh + hy)) * 16 +
                        (2 * ww + wx)];
    mval *= 0.125f;
    a0 += wfc[flat] * mval;
    a1 += wfc[9216 + flat] * mval;
  }
  s0[tid] = a0;
  s1[tid] = a1;
  __syncthreads();
  for (int st = 128; st > 0; st >>= 1) {
    if (tid < st) { s0[tid] += s0[tid + st]; s1[tid] += s1[tid + st]; }
    __syncthreads();
  }
  if (tid == 0) {
    out[b * 2 + 0] = s0[0] + bfc[0];
    out[b * 2 + 1] = s1[0] + bfc[1];
  }
}

// ---------------------------------------------------------------------------
// generic transposed conv, stride 2, kernel 2 (no overlap): out channel-last
// wt layout (Cin, O, 2,2,2)
// ---------------------------------------------------------------------------
__global__ void convT_kernel(const float* __restrict__ in, const float* __restrict__ wt,
                             const float* __restrict__ bt, float* __restrict__ out, int Cin,
                             int O, int Din, int Hin, int Win, int in_cl) {
  int i = blockIdx.x * 256 + threadIdx.x;
  int o = i % O;
  int t = i / O;
  int ww = t % (2 * Win); t /= (2 * Win);
  int hh = t % (2 * Hin); t /= (2 * Hin);
  int dd = t % (2 * Din);
  int b = t / (2 * Din);
  int sd = dd >> 1, sh = hh >> 1, sw = ww >> 1;
  int widx = (dd & 1) * 4 + (hh & 1) * 2 + (ww & 1);
  int DHW = Din * Hin * Win;
  int sp = (sd * Hin + sh) * Win + sw;
  float acc = bt[o];
  for (int c = 0; c < Cin; ++c) {
    float iv = in_cl ? in[((size_t)b * DHW + sp) * Cin + c]
                     : in[(size_t)(b * Cin + c) * DHW + sp];
    acc += iv * wt[((size_t)c * O + o) * 8 + widx];
  }
  out[i] = acc;
}

// ---------------------------------------------------------------------------
extern "C" void kernel_launch(void* const* d_in, const int* in_sizes, int n_in, void* d_out,
                              int out_size, void* d_ws, size_t ws_size, hipStream_t stream) {
  (void)in_sizes; (void)n_in; (void)out_size;
  if (ws_size < (size_t)WS_FLOATS * sizeof(float)) return;  // workspace too small

  const float* x      = (const float*)d_in[0];
  const float* w_off1 = (const float*)d_in[1];
  const float* b_off1 = (const float*)d_in[2];
  const float* w1     = (const float*)d_in[3];
  const float* b1     = (const float*)d_in[4];
  const float* g1     = (const float*)d_in[5];
  const float* be1    = (const float*)d_in[6];
  const float* w_off2 = (const float*)d_in[7];
  const float* b_off2 = (const float*)d_in[8];
  const float* w2     = (const float*)d_in[9];
  const float* b2     = (const float*)d_in[10];
  const float* g2     = (const float*)d_in[11];
  const float* be2    = (const float*)d_in[12];
  const float* wt1    = (const float*)d_in[13];
  const float* bt1    = (const float*)d_in[14];
  const float* g3     = (const float*)d_in[15];
  const float* be3    = (const float*)d_in[16];
  const float* wt2    = (const float*)d_in[17];
  const float* bt2    = (const float*)d_in[18];
  const float* g4     = (const float*)d_in[19];
  const float* be4    = (const float*)d_in[20];
  const float* wfc    = (const float*)d_in[21];
  const float* bfc    = (const float*)d_in[22];

  float* ws = (float*)d_ws;
  float* h1pre = ws + OFF_H1PRE;
  float* off2  = ws + OFF_OFF2;
  float* h2    = ws + OFF_H2;
  float* d1p   = ws + OFF_D1P;
  float* d1a   = ws + OFF_D1A;
  float* d2p   = ws + OFF_D2P;
  float* h1p   = ws + OFF_H1P;
  float* stats = ws + OFF_STATS;
  float* S1 = stats;        // 64
  float* S2 = stats + 64;   // 128
  float* S3 = stats + 192;  // 64
  float* S4 = stats + 256;  // 2

  float* dout    = (float*)d_out;
  float* o_codes = dout;            // 589824 (8,64,6,12,16) NCDHW
  float* o_dec   = dout + 589824;   // 589824 (8,1,24,48,64)
  float* o_cls   = dout + 1179648;  // 16

  (void)hipMemsetAsync(stats, 0, 512 * sizeof(float), stream);

  // ---- encoder layer 1 ----
  deform1_kernel<<<NV1 / 256, 256, 0, stream>>>(x, w_off1, b_off1, w1, b1, h1pre);
  stats_kernel<<<512, 256, 0, stream>>>(h1pre, NV1, 32, S1);
  bn_elu_pool_kernel<<<(NR2 * 32) / 256, 256, 0, stream>>>(
      h1pre, h1p, nullptr, D1_, H1_, W1_, 32, S1, g1, be1, 1.f / (float)NV1);

  // ---- encoder layer 2 (WMMA) ----
  offgemm2_kernel<<<NR2 / 16, 256, 0, stream>>>(h1p, w_off2, b_off2, off2);
  deformgemm2_kernel<<<NR2 / 32, 256, 0, stream>>>(h1p, off2, w2, b2, h2);
  stats_kernel<<<512, 256, 0, stream>>>(h2, NR2, 64, S2);
  bn_elu_pool_kernel<<<589824 / 256, 256, 0, stream>>>(
      h2, nullptr, o_codes, D2_, H2_, W2_, 64, S2, g2, be2, 1.f / (float)NR2);

  // ---- classifier head ----
  fc_kernel<<<B_, 256, 0, stream>>>(o_codes, wfc, bfc, o_cls);

  // ---- decoder ----
  convT_kernel<<<(NR2 * 32) / 256, 256, 0, stream>>>(o_codes, wt1, bt1, d1p, 64, 32, 6, 12, 16, 0);
  stats_kernel<<<512, 256, 0, stream>>>(d1p, NR2, 32, S3);
  bn_elu_kernel<<<(NR2 * 32) / 256, 256, 0, stream>>>(d1p, d1a, NR2 * 32, 32, S3, g3, be3,
                                                      1.f / (float)NR2);
  convT_kernel<<<NV1 / 256, 256, 0, stream>>>(d1a, wt2, bt2, d2p, 32, 1, 12, 24, 32, 1);
  stats_kernel<<<512, 256, 0, stream>>>(d2p, NV1, 1, S4);
  bn_elu_kernel<<<NV1 / 256, 256, 0, stream>>>(d2p, o_dec, NV1, 1, S4, g4, be4,
                                               1.f / (float)NV1);
}
